// Encoder_block_43516608643189
// MI455X (gfx1250) — matrix operand order repaired and checked by reading, not compiled, not run on hardware
//
#include <hip/hip_runtime.h>
#include <cstdint>
#include <cstddef>

typedef __attribute__((ext_vector_type(16))) __bf16 v16bf;
typedef __attribute__((ext_vector_type(8)))  __bf16 v8bf;
typedef __attribute__((ext_vector_type(4)))  __bf16 v4bf;
typedef __attribute__((ext_vector_type(8)))  float  v8f;
typedef __attribute__((ext_vector_type(4)))  float  v4f;
typedef __attribute__((ext_vector_type(4)))  int    v4i;

#define DIM 768
#define HEADS 12
#define HD 64
#define HIDDEN 3072
#define SEQ 2048
#define BATCH 4
#define MROWS (BATCH*SEQ)

#define AS1 __attribute__((address_space(1)))
#define AS3 __attribute__((address_space(3)))

#if defined(__has_builtin)
#if __has_builtin(__builtin_amdgcn_global_load_async_to_lds_b128)
#define HAVE_ASYNC_LDS 1
#endif
#endif

__device__ inline float gelu_f(float x){
  return 0.5f * x * (1.0f + erff(x * 0.70710678118654752f));
}

#if defined(HAVE_ASYNC_LDS)
// 16-byte async DMA: global -> LDS, tracked by ASYNCcnt (no VGPR round trip,
// unordered with the DS pipe so it does not stall fragment ds_loads).
// Prototype (from clang diagnostic): (int4 AS1*, int4 AS3*, i32 offset, i32 cpol)
__device__ inline void async_ld16(const __bf16* g, __bf16* l){
  __builtin_amdgcn_global_load_async_to_lds_b128(
      (AS1 v4i*)(g), (AS3 v4i*)(l), 0, 0);
}
__device__ inline void wait_async0(){
#if __has_builtin(__builtin_amdgcn_s_wait_asynccnt)
  __builtin_amdgcn_s_wait_asynccnt(0);
#else
  asm volatile("s_wait_asynccnt 0" ::: "memory");
#endif
}
#endif

// A-operand fragment (16x32 bf16, MxK): lane covers row M = lane&15,
// K = 0..15 for lanes 0-15, K = 16..31 for lanes 16-31, contiguous.
__device__ inline v16bf frag_a(const __bf16* rowp, int lane){
  const __bf16* p0 = rowp + ((lane & 16) ? 16 : 0);
  v8bf c0 = *(const v8bf*)(p0);
  v8bf c1 = *(const v8bf*)(p0 + 8);
  v16bf r;
#pragma unroll
  for (int i = 0; i < 8; ++i){ r[i] = c0[i]; r[i + 8] = c1[i]; }
  return r;
}

// B-operand fragment (32x16 bf16, KxN): lane covers column N = lane&15,
// K = 0..15 for lanes 0-15, K = 16..31 for lanes 16-31, contiguous.
__device__ inline v16bf frag_b(const __bf16* rowp, int lane){
  const __bf16* p = rowp + ((lane & 16) ? 16 : 0);
  v8bf c0 = *(const v8bf*)(p);
  v8bf c1 = *(const v8bf*)(p + 8);
  v16bf r;
#pragma unroll
  for (int i = 0; i < 8; ++i){ r[i] = c0[i]; r[i + 8] = c1[i]; }
  return r;
}

__device__ inline v8f wmma_bf16(v16bf a, v16bf b, v8f c){
  return __builtin_amdgcn_wmma_f32_16x16x32_bf16(
      /*neg_a=*/false, a, /*neg_b=*/false, b,
      /*c_mod=*/(short)0, c, /*reuse_a=*/false, /*reuse_b=*/false);
}

// ---------------------------------------------------------------------------
// fp32 -> bf16 bulk convert (4 elems / thread)
// ---------------------------------------------------------------------------
__global__ __launch_bounds__(256) void cvt_f32_bf16(
    const float* __restrict__ in, __bf16* __restrict__ out, int n4){
  int i = blockIdx.x * 256 + threadIdx.x;
  if (i < n4){
    v4f v = *(const v4f*)(in + 4 * (size_t)i);
    v4bf o;
    o[0] = (__bf16)v[0]; o[1] = (__bf16)v[1];
    o[2] = (__bf16)v[2]; o[3] = (__bf16)v[3];
    *(v4bf*)(out + 4 * (size_t)i) = o;
  }
}

// ---------------------------------------------------------------------------
// LayerNorm: one 256-thread block per row of 768, fp32 in -> bf16 out
// ---------------------------------------------------------------------------
__global__ __launch_bounds__(256) void layernorm_bf16_k(
    const float* __restrict__ x, const float* __restrict__ w,
    const float* __restrict__ bb, __bf16* __restrict__ out){
  __shared__ float red[256];
  const int row = blockIdx.x, t = threadIdx.x;
  const float* xr = x + (size_t)row * DIM;
  float v[3];
  float s = 0.f;
#pragma unroll
  for (int j = 0; j < 3; ++j){ v[j] = xr[t + j * 256]; s += v[j]; }
  red[t] = s; __syncthreads();
  for (int off = 128; off > 0; off >>= 1){
    if (t < off) red[t] += red[t + off];
    __syncthreads();
  }
  const float mu = red[0] * (1.0f / DIM);
  __syncthreads();
  float ss = 0.f;
#pragma unroll
  for (int j = 0; j < 3; ++j){ float d = v[j] - mu; ss += d * d; }
  red[t] = ss; __syncthreads();
  for (int off = 128; off > 0; off >>= 1){
    if (t < off) red[t] += red[t + off];
    __syncthreads();
  }
  const float rstd = rsqrtf(red[0] * (1.0f / DIM) + 1e-5f);
#pragma unroll
  for (int j = 0; j < 3; ++j){
    int idx = t + j * 256;
    out[(size_t)row * DIM + idx] = (__bf16)((v[j] - mu) * rstd * w[idx] + bb[idx]);
  }
}

// ---------------------------------------------------------------------------
// GEMM: out[m,n] = sum_k A[m,k]*W[n,k]  (A: MxK bf16, W: NxK bf16, both K-major)
// Block tile 128x128, BK=32, 8 waves (4x2), wave tile 32x64, double-buffered LDS.
// Staging uses GLOBAL_LOAD_ASYNC_TO_LDS_B128 (ASYNCcnt) when available, else
// register-staged software pipelining (loads -> compute -> ds_store -> barrier).
// mode 0: split qkv -> q/k head-major + v transposed (bf16)
// mode 1: fp32 out = resid + acc + bias
// mode 2: bf16 out = gelu(acc + bias)
// mode 3: fp32 out = resid + gelu(acc + bias)
// ---------------------------------------------------------------------------
__global__ __launch_bounds__(256) void gemm_bf16_k(
    const __bf16* __restrict__ A, const __bf16* __restrict__ W,
    int N, int K, int mode,
    const float* __restrict__ bias, const float* __restrict__ resid,
    float* __restrict__ outF, __bf16* __restrict__ outB,
    __bf16* __restrict__ qb, __bf16* __restrict__ kbuf, __bf16* __restrict__ vtb)
{
  const int LS = 40;                    // LDS row stride in bf16 (80B, 16B-aligned chunks)
  __shared__ __bf16 As[2][128 * 40];
  __shared__ __bf16 Bs[2][128 * 40];
  const int n0 = blockIdx.x * 128, m0 = blockIdx.y * 128;
  const int t = threadIdx.x, lane = t & 31, wid = t >> 5;
  const int wm = (wid & 3) * 32, wn = (wid >> 2) * 64;
  const int lcol = lane & 15, lhalf = lane >> 4;
  const int row0 = t >> 2, kc0 = (t & 3) * 8;   // this thread's two 16B chunks: rows row0, row0+64

  v8f c[2][4];
#pragma unroll
  for (int i = 0; i < 2; ++i)
#pragma unroll
    for (int j = 0; j < 4; ++j) c[i][j] = (v8f)(0.f);

  auto compute = [&](int cur){
    v16bf af[2], bfm[4];
#pragma unroll
    for (int mt = 0; mt < 2; ++mt)
      af[mt] = frag_a(&As[cur][(wm + mt * 16 + lcol) * LS], lane);
#pragma unroll
    for (int nt = 0; nt < 4; ++nt)
      bfm[nt] = frag_b(&Bs[cur][(wn + nt * 16 + lcol) * LS], lane);
#pragma unroll
    for (int mt = 0; mt < 2; ++mt)
#pragma unroll
      for (int nt = 0; nt < 4; ++nt)
        c[mt][nt] = wmma_bf16(af[mt], bfm[nt], c[mt][nt]);
  };

  const int KT = K / 32;

#if defined(HAVE_ASYNC_LDS)
  auto issue_tile = [&](int buf, int kt){
    const int kbase = kt * 32;
#pragma unroll
    for (int i = 0; i < 2; ++i){
      const int row = row0 + i * 64;
      async_ld16(A + (size_t)(m0 + row) * K + kbase + kc0, &As[buf][row * LS + kc0]);
      async_ld16(W + (size_t)(n0 + row) * K + kbase + kc0, &Bs[buf][row * LS + kc0]);
    }
  };
  issue_tile(0, 0);
  wait_async0();
  __syncthreads();
  for (int kt = 0; kt < KT; ++kt){
    const int cur = kt & 1;
    if (kt + 1 < KT) issue_tile(cur ^ 1, kt + 1);
    compute(cur);
    if (kt + 1 < KT) wait_async0();
    __syncthreads();
  }
#else
  v8bf rA[2], rB[2];
  auto gload_tile = [&](int kt){
    const int kbase = kt * 32;
#pragma unroll
    for (int i = 0; i < 2; ++i){
      const int row = row0 + i * 64;
      rA[i] = *(const v8bf*)(A + (size_t)(m0 + row) * K + kbase + kc0);
      rB[i] = *(const v8bf*)(W + (size_t)(n0 + row) * K + kbase + kc0);
    }
  };
  auto commit_tile = [&](int buf){
#pragma unroll
    for (int i = 0; i < 2; ++i){
      const int row = row0 + i * 64;
      *(v8bf*)(&As[buf][row * LS + kc0]) = rA[i];
      *(v8bf*)(&Bs[buf][row * LS + kc0]) = rB[i];
    }
  };
  gload_tile(0);
  commit_tile(0);
  __syncthreads();
  for (int kt = 0; kt < KT; ++kt){
    const int cur = kt & 1;
    if (kt + 1 < KT) gload_tile(kt + 1);     // global loads overlap compute
    compute(cur);
    if (kt + 1 < KT) commit_tile(cur ^ 1);   // ds_stores after frag ds_loads
    __syncthreads();
  }
#endif

  // Epilogue. C layout: row M = r + 8*lhalf (per v8f element r), col N = lcol.
#pragma unroll
  for (int mt = 0; mt < 2; ++mt){
#pragma unroll
    for (int nt = 0; nt < 4; ++nt){
#pragma unroll
      for (int r = 0; r < 8; ++r){
        const int m = m0 + wm + mt * 16 + lhalf * 8 + r;
        const int n = n0 + wn + nt * 16 + lcol;
        float v = c[mt][nt][r];
        if (mode == 0){
          const int which = n / DIM, rem = n % DIM;
          const int head = rem / HD, d = rem % HD;
          const int b = m / SEQ, tok = m % SEQ;
          const size_t bh = (size_t)b * HEADS + head;
          if (which == 0)      qb  [(bh * SEQ + tok) * HD + d] = (__bf16)v;
          else if (which == 1) kbuf[(bh * SEQ + tok) * HD + d] = (__bf16)v;
          else                 vtb [(bh * HD + d) * SEQ + tok] = (__bf16)v;
        } else if (mode == 1){
          const size_t idx = (size_t)m * DIM + n;
          outF[idx] = resid[idx] + v + bias[n];
        } else if (mode == 2){
          outB[(size_t)m * (size_t)N + n] = (__bf16)gelu_f(v + bias[n]);
        } else {
          const size_t idx = (size_t)m * DIM + n;
          outF[idx] = resid[idx] + gelu_f(v + bias[n]);
        }
      }
    }
  }
}

// ---------------------------------------------------------------------------
// Flash attention: 1 wave = 16 Q rows, 4 waves/block, block = (b,h, 64 q rows).
// S = Q K^T * 1/8, online softmax, O = P V. P tile (16x32) is converted from
// WMMA C-layout to A-layout through a private LDS slice (same-wave, in-order).
// ---------------------------------------------------------------------------
__global__ __launch_bounds__(128) void attn_k(
    const __bf16* __restrict__ q, const __bf16* __restrict__ k,
    const __bf16* __restrict__ vt, __bf16* __restrict__ o)
{
  __shared__ __bf16 pS[4][16 * 40];
  const int lane = threadIdx.x & 31, wid = threadIdx.x >> 5;
  const int lcol = lane & 15, lhalf = lane >> 4;
  const int b = blockIdx.z, h = blockIdx.y;
  const size_t bh = (size_t)b * HEADS + h;
  const int q0 = blockIdx.x * 64 + wid * 16;
  const __bf16* Q  = q  + bh * SEQ * HD;
  const __bf16* Kp = k  + bh * SEQ * HD;
  const __bf16* VT = vt + bh * HD * SEQ;

  v16bf aq[2];
#pragma unroll
  for (int kk = 0; kk < 2; ++kk)
    aq[kk] = frag_a(Q + (size_t)(q0 + lcol) * HD + kk * 32, lane);

  v8f oacc[4];
#pragma unroll
  for (int i = 0; i < 4; ++i) oacc[i] = (v8f)(0.f);
  float mrow[8], lrow[8];
#pragma unroll
  for (int r = 0; r < 8; ++r){ mrow[r] = -3.0e38f; lrow[r] = 0.f; }

  __bf16* pw = pS[wid];

  for (int kt0 = 0; kt0 < SEQ; kt0 += 32){
    v8f s[2];
    s[0] = (v8f)(0.f); s[1] = (v8f)(0.f);
#pragma unroll
    for (int tt = 0; tt < 2; ++tt)
#pragma unroll
      for (int kk = 0; kk < 2; ++kk){
        v16bf bk = frag_b(Kp + (size_t)(kt0 + tt * 16 + lcol) * HD + kk * 32, lane);
        s[tt] = wmma_bf16(aq[kk], bk, s[tt]);
      }

    const float sc = 0.125f;              // 1/sqrt(64)
    float p0a[8], p1a[8];
#pragma unroll
    for (int r = 0; r < 8; ++r){
      float v0 = s[0][r] * sc, v1 = s[1][r] * sc;
      float mloc = fmaxf(v0, v1);
#pragma unroll
      for (int off = 1; off < 16; off <<= 1)
        mloc = fmaxf(mloc, __shfl_xor(mloc, off));
      const float mnew  = fmaxf(mrow[r], mloc);
      const float alpha = __expf(mrow[r] - mnew);
      const float p0 = __expf(v0 - mnew), p1 = __expf(v1 - mnew);
      float sloc = p0 + p1;
#pragma unroll
      for (int off = 1; off < 16; off <<= 1)
        sloc += __shfl_xor(sloc, off);
      lrow[r] = lrow[r] * alpha + sloc;
      mrow[r] = mnew;
#pragma unroll
      for (int nt = 0; nt < 4; ++nt) oacc[nt][r] *= alpha;
      p0a[r] = p0; p1a[r] = p1;
    }

    // C-layout -> LDS (natural m,n) -> reload in A-layout
#pragma unroll
    for (int r = 0; r < 8; ++r){
      pw[(lhalf * 8 + r) * 40 + lcol]      = (__bf16)p0a[r];
      pw[(lhalf * 8 + r) * 40 + 16 + lcol] = (__bf16)p1a[r];
    }
    asm volatile("" ::: "memory");        // same-wave LDS: hardware in-order
    v16bf pa = frag_a(pw + lcol * 40, lane);
#pragma unroll
    for (int nt = 0; nt < 4; ++nt){
      v16bf bv = frag_b(VT + (size_t)(nt * 16 + lcol) * SEQ + kt0, lane);
      oacc[nt] = wmma_bf16(pa, bv, oacc[nt]);
    }
  }

#pragma unroll
  for (int r = 0; r < 8; ++r){
    const float inv = 1.0f / lrow[r];
    const int tok = q0 + lhalf * 8 + r;
#pragma unroll
    for (int nt = 0; nt < 4; ++nt){
      const int d = nt * 16 + lcol;
      o[((size_t)(b * SEQ + tok)) * DIM + h * HD + d] = (__bf16)(oacc[nt][r] * inv);
    }
  }
}

// ---------------------------------------------------------------------------
extern "C" void kernel_launch(void* const* d_in, const int* in_sizes, int n_in,
                              void* d_out, int out_size, void* d_ws, size_t ws_size,
                              hipStream_t stream){
  (void)in_sizes; (void)n_in; (void)out_size; (void)ws_size;
  const float* x      = (const float*)d_in[0];
  const float* ln_w   = (const float*)d_in[1];
  const float* ln_b   = (const float*)d_in[2];
  const float* qkv_w  = (const float*)d_in[3];
  const float* proj_w = (const float*)d_in[4];
  const float* proj_b = (const float*)d_in[5];
  const float* fc1_w  = (const float*)d_in[6];
  const float* fc1_b  = (const float*)d_in[7];
  const float* fc2_w  = (const float*)d_in[8];
  const float* fc2_b  = (const float*)d_in[9];
  float* out = (float*)d_out;

  char* ws = (char*)d_ws;
  size_t off = 0;
  auto alloc = [&](size_t bytes) -> char* {
    char* p = ws + off;
    off = (off + bytes + 255) & ~(size_t)255;
    return p;
  };
  __bf16* wq = (__bf16*)alloc((size_t)3 * DIM * DIM * 2);
  __bf16* wp = (__bf16*)alloc((size_t)DIM * DIM * 2);
  __bf16* w1 = (__bf16*)alloc((size_t)HIDDEN * DIM * 2);
  __bf16* w2 = (__bf16*)alloc((size_t)DIM * HIDDEN * 2);
  __bf16* h1 = (__bf16*)alloc((size_t)MROWS * DIM * 2);
  __bf16* qb = (__bf16*)alloc((size_t)MROWS * DIM * 2);
  __bf16* kb = (__bf16*)alloc((size_t)MROWS * DIM * 2);
  __bf16* vt = (__bf16*)alloc((size_t)MROWS * DIM * 2);
  __bf16* ob = (__bf16*)alloc((size_t)MROWS * DIM * 2);
  float*  x2 = (float*) alloc((size_t)MROWS * DIM * 4);
  __bf16* h2 = (__bf16*)alloc((size_t)MROWS * DIM * 2);
  __bf16* hf = (__bf16*)alloc((size_t)MROWS * HIDDEN * 2);

  auto cvt = [&](const float* src, __bf16* dst, size_t n){
    int n4 = (int)(n / 4);
    cvt_f32_bf16<<<(n4 + 255) / 256, 256, 0, stream>>>(src, dst, n4);
  };
  cvt(qkv_w, wq, (size_t)3 * DIM * DIM);
  cvt(proj_w, wp, (size_t)DIM * DIM);
  cvt(fc1_w, w1, (size_t)HIDDEN * DIM);
  cvt(fc2_w, w2, (size_t)DIM * HIDDEN);

  // 1) h1 = LN(x)
  layernorm_bf16_k<<<MROWS, 256, 0, stream>>>(x, ln_w, ln_b, h1);
  // 2) qkv = h1 @ qkv_w^T  -> q/k/vT buffers
  gemm_bf16_k<<<dim3(3 * DIM / 128, MROWS / 128), 256, 0, stream>>>(
      h1, wq, 3 * DIM, DIM, 0, nullptr, nullptr, nullptr, nullptr, qb, kb, vt);
  // 3) attention -> ob
  attn_k<<<dim3(SEQ / 64, HEADS, BATCH), 128, 0, stream>>>(qb, kb, vt, ob);
  // 4) x2 = x + ob @ proj_w^T + proj_b
  gemm_bf16_k<<<dim3(DIM / 128, MROWS / 128), 256, 0, stream>>>(
      ob, wp, DIM, DIM, 1, proj_b, x, x2, nullptr, nullptr, nullptr, nullptr);
  // 5) h2 = LN(x2)
  layernorm_bf16_k<<<MROWS, 256, 0, stream>>>(x2, ln_w, ln_b, h2);
  // 6) hf = gelu(h2 @ fc1_w^T + fc1_b)
  gemm_bf16_k<<<dim3(HIDDEN / 128, MROWS / 128), 256, 0, stream>>>(
      h2, w1, HIDDEN, DIM, 2, fc1_b, nullptr, nullptr, hf, nullptr, nullptr, nullptr);
  // 7) out = x2 + gelu(hf @ fc2_w^T + fc2_b)
  gemm_bf16_k<<<dim3(DIM / 128, MROWS / 128), 256, 0, stream>>>(
      hf, w2, DIM, HIDDEN, 3, fc2_b, x2, out, nullptr, nullptr, nullptr, nullptr);
}
